// Attention_84086869721433
// MI455X (gfx1250) — matrix operand order repaired and checked by reading, not compiled, not run on hardware
//
#include <hip/hip_runtime.h>
#include <hip/hip_bf16.h>

// ---------------------------------------------------------------------------
// Attention block for MI455X (gfx1250, wave32, WMMA).
//   B=8, N=1024, C=768, H=12, HD=64, SCALE=0.125
// Pipeline:
//   0) cvt_kernel   : x f32 -> bf16 (row-major)
//      cvtT_kernel  : w_qkv, w_proj f32 [K][N] -> bf16 [N][K] (one-time transpose)
//   1) gemm<0>      : xb @ wqkvT -> q,k,v (bf16 [B*H][N][64]), q pre-scaled
//   2) trV_kernel   : v -> vT (bf16 [B*H][64][N]) one-time per-head transpose
//   3) attn_kernel  : flash attention per (b,h) -> ctx (bf16 [B][N][C])
//                     8 waves/block, 128 queries, shared K/V staging
//   4) gemm<1>      : ctx @ wprojT + bias -> d_out (f32)
// All matrix math via v_wmma_f32_16x16x32_bf16, f32 accumulate.
// Hot-loop staging is pure b128 copy; LDS double-buffered.
// ---------------------------------------------------------------------------

#define BATCH 8
#define SEQ   1024
#define DIM   768
#define HEADS 12
#define HDIM  64

typedef __bf16 v16bf __attribute__((ext_vector_type(16)));
typedef float  v8f   __attribute__((ext_vector_type(8)));

__device__ __forceinline__ unsigned short f2bf(float f) {
    unsigned int u = __builtin_bit_cast(unsigned int, f);
    u += 0x7fffu + ((u >> 16) & 1u);           // round-to-nearest-even
    return (unsigned short)(u >> 16);
}

__device__ __forceinline__ v16bf load_frag16(const unsigned short* p0,
                                             const unsigned short* p1) {
    v16bf f;
    uint4* u = reinterpret_cast<uint4*>(&f);
    u[0] = *reinterpret_cast<const uint4*>(p0);
    u[1] = *reinterpret_cast<const uint4*>(p1);
    return f;
}

__device__ __forceinline__ v8f wmma_bf16(v16bf a, v16bf b, v8f c) {
    return __builtin_amdgcn_wmma_f32_16x16x32_bf16(
        false, a, false, b, (short)0, c, false, false);
}

// ---------------------------------------------------------------------------
// f32 -> bf16, row-major (for x).
// ---------------------------------------------------------------------------
__global__ __launch_bounds__(256) void cvt_kernel(
    const float* __restrict__ in, unsigned short* __restrict__ out, int n4)
{
    int i = blockIdx.x * 256 + threadIdx.x;
    if (i < n4) {
        float4 f = reinterpret_cast<const float4*>(in)[i];
        union { unsigned short s[4]; uint2 u; } p;
        p.s[0] = f2bf(f.x); p.s[1] = f2bf(f.y);
        p.s[2] = f2bf(f.z); p.s[3] = f2bf(f.w);
        reinterpret_cast<uint2*>(out)[i] = p.u;
    }
}

// ---------------------------------------------------------------------------
// f32 [K][N] -> bf16 [N][K], 64x64 tiles via LDS (one-time, for weights).
// grid = (K/64, N/64), 256 threads.
// ---------------------------------------------------------------------------
__global__ __launch_bounds__(256) void cvtT_kernel(
    const float* __restrict__ in, unsigned short* __restrict__ out, int K, int N)
{
    __shared__ __align__(16) unsigned short Ts[64][80];   // [n][k], padded
    const int tid = threadIdx.x;
    const int k0 = blockIdx.x * 64, n0 = blockIdx.y * 64;
#pragma unroll
    for (int i = 0; i < 4; ++i) {              // 1024 float4, 4/thread
        int vno = tid + i * 256;
        int row = vno >> 4;                    // k
        int c4  = (vno & 15) * 4;              // n
        float4 f = *reinterpret_cast<const float4*>(in + (size_t)(k0 + row) * N + n0 + c4);
        Ts[c4 + 0][row] = f2bf(f.x);
        Ts[c4 + 1][row] = f2bf(f.y);
        Ts[c4 + 2][row] = f2bf(f.z);
        Ts[c4 + 3][row] = f2bf(f.w);
    }
    __syncthreads();
#pragma unroll
    for (int i = 0; i < 2; ++i) {              // 512 uint4, 2/thread
        int vno = tid + i * 256;
        int n  = vno >> 3;
        int c8 = (vno & 7) * 8;
        *reinterpret_cast<uint4*>(out + (size_t)(n0 + n) * K + k0 + c8) =
            *reinterpret_cast<const uint4*>(&Ts[n][c8]);
    }
}

// ---------------------------------------------------------------------------
// v bf16 [bh][SEQ][64] -> vT bf16 [bh][64][SEQ], coalesced both sides.
// grid = (SEQ/64, B*H), 256 threads.
// ---------------------------------------------------------------------------
__global__ __launch_bounds__(256) void trV_kernel(
    const unsigned short* __restrict__ v, unsigned short* __restrict__ vT)
{
    __shared__ __align__(16) unsigned short Ts[64][80];   // [d][key], padded
    const int tid = threadIdx.x;
    const int key0 = blockIdx.x * 64;
    const size_t ho = (size_t)blockIdx.y * SEQ * HDIM;
#pragma unroll
    for (int i = 0; i < 2; ++i) {              // 512 uint4 in, 2/thread
        int vno = tid + i * 256;
        int row = vno >> 3;                    // key
        int c8  = (vno & 7) * 8;               // d
        union { uint4 u; unsigned short s[8]; } w;
        w.u = *reinterpret_cast<const uint4*>(v + ho + (size_t)(key0 + row) * HDIM + c8);
#pragma unroll
        for (int j = 0; j < 8; ++j) Ts[c8 + j][row] = w.s[j];
    }
    __syncthreads();
#pragma unroll
    for (int i = 0; i < 2; ++i) {              // 512 uint4 out, 2/thread
        int vno = tid + i * 256;
        int d  = vno >> 3;
        int c8 = (vno & 7) * 8;
        *reinterpret_cast<uint4*>(vT + ho + (size_t)d * SEQ + key0 + c8) =
            *reinterpret_cast<const uint4*>(&Ts[d][c8]);
    }
}

// ---------------------------------------------------------------------------
// GEMM: D[M,N] = A[M,K] (bf16, row-major) * Bt[N,K] (bf16, pre-transposed).
// Block 256 thr = 8 waves; tile 128x64, K-tile 64; double-buffered LDS.
// EPI==0: scatter q/k/v bf16 [B*H][N][64], q scaled 0.125
// EPI==1: out = acc + bias (f32)
// ---------------------------------------------------------------------------
template <int EPI>
__global__ __launch_bounds__(256) void gemm_kernel(
    const unsigned short* __restrict__ A, const unsigned short* __restrict__ Bt,
    int M, int N, int K,
    unsigned short* __restrict__ qd, unsigned short* __restrict__ kd,
    unsigned short* __restrict__ vd,
    const float* __restrict__ bias, float* __restrict__ out)
{
    __shared__ __align__(16) unsigned short As[2][128][64];   // 32KB
    __shared__ __align__(16) unsigned short Bs[2][64][64];    // 16KB

    const int tid  = threadIdx.x;
    const int lane = tid & 31;
    const int wid  = tid >> 5;
    const int wm   = (wid & 3) * 32;
    const int wn   = (wid >> 2) * 32;
    const int g    = lane >> 4;
    const int l16  = lane & 15;
    const int gm0  = blockIdx.x * 128;
    const int gn0  = blockIdx.y * 64;

    uint4 ra[4], rb[2];
    auto fetch = [&](int kt) {
        const int k0 = kt << 6;
#pragma unroll
        for (int i = 0; i < 4; ++i) {
            int vno = tid + i * 256, row = vno >> 3, c8 = (vno & 7) * 8;
            ra[i] = *reinterpret_cast<const uint4*>(A + (size_t)(gm0 + row) * K + k0 + c8);
        }
#pragma unroll
        for (int i = 0; i < 2; ++i) {
            int vno = tid + i * 256, n = vno >> 3, c8 = (vno & 7) * 8;
            rb[i] = *reinterpret_cast<const uint4*>(Bt + (size_t)(gn0 + n) * K + k0 + c8);
        }
    };
    auto commit = [&](int buf) {
#pragma unroll
        for (int i = 0; i < 4; ++i) {
            int vno = tid + i * 256, row = vno >> 3, c8 = (vno & 7) * 8;
            *reinterpret_cast<uint4*>(&As[buf][row][c8]) = ra[i];
        }
#pragma unroll
        for (int i = 0; i < 2; ++i) {
            int vno = tid + i * 256, n = vno >> 3, c8 = (vno & 7) * 8;
            *reinterpret_cast<uint4*>(&Bs[buf][n][c8]) = rb[i];
        }
    };

    v8f acc[2][2];
#pragma unroll
    for (int i = 0; i < 2; ++i)
#pragma unroll
        for (int j = 0; j < 2; ++j)
            acc[i][j] = (v8f){0.f, 0.f, 0.f, 0.f, 0.f, 0.f, 0.f, 0.f};

    const int ksteps = K >> 6;
    fetch(0);
    commit(0);
    for (int kt = 0; kt < ksteps; ++kt) {
        __syncthreads();
        const int cur = kt & 1;
        if (kt + 1 < ksteps) fetch(kt + 1);     // overlap with WMMA below
#pragma unroll
        for (int ks = 0; ks < 64; ks += 32) {
            v16bf a0 = load_frag16(&As[cur][wm + l16][ks + 16 * g],
                                   &As[cur][wm + l16][ks + 16 * g + 8]);
            v16bf a1 = load_frag16(&As[cur][wm + 16 + l16][ks + 16 * g],
                                   &As[cur][wm + 16 + l16][ks + 16 * g + 8]);
            v16bf b0 = load_frag16(&Bs[cur][wn + l16][ks + 16 * g],
                                   &Bs[cur][wn + l16][ks + 16 * g + 8]);
            v16bf b1 = load_frag16(&Bs[cur][wn + 16 + l16][ks + 16 * g],
                                   &Bs[cur][wn + 16 + l16][ks + 16 * g + 8]);
            acc[0][0] = wmma_bf16(a0, b0, acc[0][0]);
            acc[0][1] = wmma_bf16(a0, b1, acc[0][1]);
            acc[1][0] = wmma_bf16(a1, b0, acc[1][0]);
            acc[1][1] = wmma_bf16(a1, b1, acc[1][1]);
        }
        if (kt + 1 < ksteps) commit((kt + 1) & 1);
    }

    // ---- epilogue
#pragma unroll
    for (int tm = 0; tm < 2; ++tm) {
#pragma unroll
        for (int tn = 0; tn < 2; ++tn) {
#pragma unroll
            for (int r = 0; r < 8; ++r) {
                int gm = gm0 + wm + tm * 16 + r + 8 * g;
                int gn = gn0 + wn + tn * 16 + l16;
                float val = acc[tm][tn][r];
                if constexpr (EPI == 0) {
                    int b     = gm >> 10;
                    int n     = gm & 1023;
                    int which = gn / DIM;
                    int rem   = gn - which * DIM;
                    int hh    = rem >> 6;
                    int d     = rem & 63;
                    unsigned short bv = f2bf(which == 0 ? val * 0.125f : val);
                    size_t off = (((size_t)b * HEADS + hh) * SEQ + n) * HDIM + d;
                    unsigned short* dst = (which == 0) ? qd : (which == 1) ? kd : vd;
                    dst[off] = bv;
                } else {
                    out[(size_t)gm * N + gn] = val + bias[gn];
                }
            }
        }
    }
}

// ---------------------------------------------------------------------------
// Flash attention. Block 256 thr = 8 waves; wave owns 16 queries (q-tile 128).
// 64-key chunks, double-buffered K/V^T staging shared by all 8 waves.
// 16 WMMAs per chunk per wave. Writes ctx bf16 [B][N][C].
// ---------------------------------------------------------------------------
__global__ __launch_bounds__(256) void attn_kernel(
    const unsigned short* __restrict__ q, const unsigned short* __restrict__ k,
    const unsigned short* __restrict__ vT, const float* __restrict__ mask,
    unsigned short* __restrict__ ctx)
{
    __shared__ __align__(16) unsigned short Ks[2][64][64];    // (key,d)  16KB
    __shared__ __align__(16) unsigned short Vts[2][64][64];   // (d,key)  16KB
    __shared__ __align__(16) unsigned short Ps[8][16][64];    // P stage  16KB

    const int tid  = threadIdx.x;
    const int lane = tid & 31;
    const int wid  = tid >> 5;                 // 0..7
    const int g    = lane >> 4;
    const int l16  = lane & 15;

    const int bh = blockIdx.y;
    const int b  = bh / HEADS;
    const int h  = bh - b * HEADS;
    const size_t head_off = (size_t)bh * SEQ * HDIM;
    const int q0 = blockIdx.x * 128 + wid * 16;   // this wave's query base

    uint4 rk[2], rv[2];
    auto fetch = [&](int kc) {
        const int key0 = kc * 64;
#pragma unroll
        for (int i = 0; i < 2; ++i) {          // 512 uint4 each, 2/thread
            int vno = tid + i * 256, row = vno >> 3, c8 = (vno & 7) * 8;
            rk[i] = *reinterpret_cast<const uint4*>(
                k + head_off + (size_t)(key0 + row) * HDIM + c8);   // row=key, c8=d
            rv[i] = *reinterpret_cast<const uint4*>(
                vT + head_off + (size_t)row * SEQ + key0 + c8);     // row=d, c8=key
        }
    };
    auto commit = [&](int buf) {
#pragma unroll
        for (int i = 0; i < 2; ++i) {
            int vno = tid + i * 256, row = vno >> 3, c8 = (vno & 7) * 8;
            *reinterpret_cast<uint4*>(&Ks[buf][row][c8])  = rk[i];
            *reinterpret_cast<uint4*>(&Vts[buf][row][c8]) = rv[i];
        }
    };

    // Q A-fragments (16x64 -> two 16x32 k-steps), loaded once
    const unsigned short* qrow = q + head_off + (size_t)(q0 + l16) * HDIM;
    v16bf qa0 = load_frag16(qrow + 16 * g,      qrow + 16 * g + 8);
    v16bf qa1 = load_frag16(qrow + 32 + 16 * g, qrow + 40 + 16 * g);

    v8f o[4];
#pragma unroll
    for (int dt = 0; dt < 4; ++dt)
        o[dt] = (v8f){0.f, 0.f, 0.f, 0.f, 0.f, 0.f, 0.f, 0.f};
    float mrow[8], lrow[8];
#pragma unroll
    for (int r = 0; r < 8; ++r) { mrow[r] = -1e30f; lrow[r] = 0.f; }

    const float* mask_b = mask + (size_t)b * SEQ * SEQ;
    const int nchunks = SEQ / 64;

    fetch(0);
    commit(0);
    for (int kc = 0; kc < nchunks; ++kc) {
        __syncthreads();
        const int cur = kc & 1;
        const int key0 = kc * 64;
        if (kc + 1 < nchunks) fetch(kc + 1);    // overlap with compute

        // ---- S = Q @ K^T : four 16x16 key tiles x two K-dim sub-steps
        v8f s[4];
#pragma unroll
        for (int t = 0; t < 4; ++t)
            s[t] = (v8f){0.f, 0.f, 0.f, 0.f, 0.f, 0.f, 0.f, 0.f};
        {
            v16bf kb[4];
#pragma unroll
            for (int t = 0; t < 4; ++t)
                kb[t] = load_frag16(&Ks[cur][t * 16 + l16][16 * g],
                                    &Ks[cur][t * 16 + l16][16 * g + 8]);
#pragma unroll
            for (int t = 0; t < 4; ++t) s[t] = wmma_bf16(qa0, kb[t], s[t]);
#pragma unroll
            for (int t = 0; t < 4; ++t)
                kb[t] = load_frag16(&Ks[cur][t * 16 + l16][32 + 16 * g],
                                    &Ks[cur][t * 16 + l16][32 + 16 * g + 8]);
#pragma unroll
            for (int t = 0; t < 4; ++t) s[t] = wmma_bf16(qa1, kb[t], s[t]);
        }

        // ---- online softmax per row (row = r + 8*g)
#pragma unroll
        for (int r = 0; r < 8; ++r) {
            int qglob = q0 + r + 8 * g;
            const float* mp = mask_b + (size_t)qglob * SEQ + key0;
            float sv[4];
#pragma unroll
            for (int t = 0; t < 4; ++t) sv[t] = s[t][r] + mp[t * 16 + l16];
            float cm = fmaxf(fmaxf(sv[0], sv[1]), fmaxf(sv[2], sv[3]));
#pragma unroll
            for (int off = 1; off < 16; off <<= 1)
                cm = fmaxf(cm, __shfl_xor(cm, off, 32));
            float mnew = fmaxf(mrow[r], cm);
            float p[4], ps = 0.f;
#pragma unroll
            for (int t = 0; t < 4; ++t) { p[t] = __expf(sv[t] - mnew); ps += p[t]; }
#pragma unroll
            for (int off = 1; off < 16; off <<= 1)
                ps += __shfl_xor(ps, off, 32);
            float corr = __expf(mrow[r] - mnew);
            lrow[r] = lrow[r] * corr + ps;
            mrow[r] = mnew;
#pragma unroll
            for (int dt = 0; dt < 4; ++dt) o[dt][r] *= corr;
#pragma unroll
            for (int t = 0; t < 4; ++t)
                Ps[wid][r + 8 * g][t * 16 + l16] = f2bf(p[t]);
        }

        // ---- O += P @ V : P 16x64 = two A-frags; V^T tile, contiguous keys
        {
            const unsigned short* prow = &Ps[wid][l16][0];
            v16bf pa0 = load_frag16(prow + 16 * g,      prow + 16 * g + 8);
            v16bf pa1 = load_frag16(prow + 32 + 16 * g, prow + 40 + 16 * g);
            v16bf vb[4];
#pragma unroll
            for (int dt = 0; dt < 4; ++dt)
                vb[dt] = load_frag16(&Vts[cur][16 * dt + l16][16 * g],
                                     &Vts[cur][16 * dt + l16][16 * g + 8]);
#pragma unroll
            for (int dt = 0; dt < 4; ++dt) o[dt] = wmma_bf16(pa0, vb[dt], o[dt]);
#pragma unroll
            for (int dt = 0; dt < 4; ++dt)
                vb[dt] = load_frag16(&Vts[cur][16 * dt + l16][32 + 16 * g],
                                     &Vts[cur][16 * dt + l16][32 + 16 * g + 8]);
#pragma unroll
            for (int dt = 0; dt < 4; ++dt) o[dt] = wmma_bf16(pa1, vb[dt], o[dt]);
        }
        if (kc + 1 < nchunks) commit((kc + 1) & 1);
    }

    // ---- finalize: O /= l, write ctx bf16 [B][N][C]
#pragma unroll
    for (int r = 0; r < 8; ++r) {
        int qglob = q0 + r + 8 * g;
        float inv_l = 1.f / lrow[r];
        unsigned short* crow = ctx + ((size_t)b * SEQ + qglob) * DIM + h * HDIM;
#pragma unroll
        for (int dt = 0; dt < 4; ++dt)
            crow[16 * dt + l16] = f2bf(o[dt][r] * inv_l);
    }
}

// ---------------------------------------------------------------------------
extern "C" void kernel_launch(void* const* d_in, const int* in_sizes, int n_in,
                              void* d_out, int out_size, void* d_ws, size_t ws_size,
                              hipStream_t stream) {
    const float* x      = (const float*)d_in[0];   // [B,N,C]
    const float* mask   = (const float*)d_in[1];   // [B,N,N]
    const float* w_qkv  = (const float*)d_in[2];   // [C,3C]
    const float* w_proj = (const float*)d_in[3];   // [C,C]
    const float* b_proj = (const float*)d_in[4];   // [C]
    float* out = (float*)d_out;

    const size_t nX     = (size_t)BATCH * SEQ * DIM;          // 6291456
    const size_t nWqkv  = (size_t)DIM * 3 * DIM;              // 1769472
    const size_t nWproj = (size_t)DIM * DIM;                  // 589824
    const size_t nHead  = (size_t)BATCH * HEADS * SEQ * HDIM; // 6291456

    unsigned short* xb     = (unsigned short*)d_ws;
    unsigned short* wqkvT  = xb + nX;        // bf16 [3C][C]
    unsigned short* wprojT = wqkvT + nWqkv;  // bf16 [C][C]
    unsigned short* qws    = wprojT + nWproj;
    unsigned short* kws    = qws + nHead;
    unsigned short* vws    = kws + nHead;
    unsigned short* vTws   = vws + nHead;    // bf16 [B*H][64][N]
    unsigned short* ctx    = vTws + nHead;   // bf16 [B,N,C]

    const int M = BATCH * SEQ;               // 8192

    // 0) one-time conversions / transposes
    cvt_kernel<<<(int)(nX / 4 + 255) / 256, 256, 0, stream>>>(x, xb, (int)(nX / 4));
    cvtT_kernel<<<dim3(DIM / 64, (3 * DIM) / 64), 256, 0, stream>>>(
        w_qkv, wqkvT, DIM, 3 * DIM);
    cvtT_kernel<<<dim3(DIM / 64, DIM / 64), 256, 0, stream>>>(
        w_proj, wprojT, DIM, DIM);

    // 1) QKV projection -> q,k,v bf16 (q pre-scaled by 0.125)
    gemm_kernel<0><<<dim3(M / 128, (3 * DIM) / 64), 256, 0, stream>>>(
        xb, wqkvT, M, 3 * DIM, DIM, qws, kws, vws, nullptr, nullptr);

    // 2) per-head V transpose
    trV_kernel<<<dim3(SEQ / 64, BATCH * HEADS), 256, 0, stream>>>(vws, vTws);

    // 3) Flash attention -> ctx (bf16), 128 queries per block
    attn_kernel<<<dim3(SEQ / 128, BATCH * HEADS), 256, 0, stream>>>(
        qws, kws, vTws, mask, ctx);

    // 4) Output projection + bias -> d_out (f32)
    gemm_kernel<1><<<dim3(M / 128, DIM / 64), 256, 0, stream>>>(
        ctx, wprojT, M, DIM, DIM, nullptr, nullptr, nullptr, b_proj, out);
}
